// MoELayer_53704271069272
// MI455X (gfx1250) — matrix operand order repaired and checked by reading, not compiled, not run on hardware
//
#include <hip/hip_runtime.h>
#include <math.h>

#define NTOK 16384
#define DIMX 512
#define HIDX 2048
#define NEXP 4
#define RANK 8

typedef __attribute__((ext_vector_type(16))) __bf16 v16bf;
typedef __attribute__((ext_vector_type(8)))  __bf16 v8bf;
typedef __attribute__((ext_vector_type(8)))  float  v8f;

__device__ __forceinline__ unsigned short f2bf(float f) {
  unsigned int u = __float_as_uint(f);
  u += 0x7FFFu + ((u >> 16) & 1u);           // round-to-nearest-even
  return (unsigned short)(u >> 16);
}
__device__ __forceinline__ float bf2f(unsigned short h) {
  return __uint_as_float(((unsigned int)h) << 16);
}

// ---------------------------------------------------------------- utilities
__global__ void cvt_bf16_kernel(const float* __restrict__ src,
                                unsigned short* __restrict__ dst, int n) {
  for (int i = blockIdx.x * blockDim.x + threadIdx.x; i < n;
       i += gridDim.x * blockDim.x)
    dst[i] = f2bf(src[i]);
}

__global__ void zero_kernel(float* __restrict__ p, int n) {
  for (int i = blockIdx.x * blockDim.x + threadIdx.x; i < n;
       i += gridDim.x * blockDim.x)
    p[i] = 0.0f;
}

// ---------------------------------------------------------------- router
// one wave (32 lanes) per token; top-2 softmax combine weights -> cw[N,4]
__global__ void router_kernel(const float* __restrict__ x,
                              const float* __restrict__ Wr,
                              const float* __restrict__ br,
                              float* __restrict__ cw) {
  const int lane = threadIdx.x & 31;
  const int wave = threadIdx.x >> 5;
  const int t = blockIdx.x * 8 + wave;
  float a0 = 0.f, a1 = 0.f, a2 = 0.f, a3 = 0.f;
  const float* xr = x + (size_t)t * DIMX;
  for (int d = lane; d < DIMX; d += 32) {
    float xv = xr[d];
    a0 += xv * Wr[0 * DIMX + d];
    a1 += xv * Wr[1 * DIMX + d];
    a2 += xv * Wr[2 * DIMX + d];
    a3 += xv * Wr[3 * DIMX + d];
  }
  for (int off = 16; off; off >>= 1) {
    a0 += __shfl_xor(a0, off, 32);
    a1 += __shfl_xor(a1, off, 32);
    a2 += __shfl_xor(a2, off, 32);
    a3 += __shfl_xor(a3, off, 32);
  }
  if (lane == 0) {
    float l[4] = {a0 + br[0], a1 + br[1], a2 + br[2], a3 + br[3]};
    int i0 = 0;
    for (int e = 1; e < 4; ++e) if (l[e] > l[i0]) i0 = e;
    int i1 = (i0 == 0) ? 1 : 0;
    for (int e = 0; e < 4; ++e)
      if (e != i0 && l[e] > l[i1]) i1 = e;
    float w0 = 1.0f / (1.0f + expf(l[i1] - l[i0]));   // softmax over top-2
    float w1 = 1.0f - w0;
    for (int e = 0; e < 4; ++e)
      cw[(size_t)t * 4 + e] = (e == i0) ? w0 : ((e == i1) ? w1 : 0.0f);
  }
}

// ---------------------------------------------------------------- LoRA proj
// P[t,r] = sum_d x[t,d] * A[r,d]   (x f32, one wave per token)
__global__ void lora_x_kernel(const float* __restrict__ x,
                              const float* __restrict__ A,
                              float* __restrict__ P, int D) {
  const int lane = threadIdx.x & 31;
  const int wave = threadIdx.x >> 5;
  const int t = blockIdx.x * 8 + wave;
  float acc[RANK] = {};
  const float* xr = x + (size_t)t * D;
  for (int d = lane; d < D; d += 32) {
    float xv = xr[d];
#pragma unroll
    for (int r = 0; r < RANK; ++r) acc[r] += xv * A[(size_t)r * D + d];
  }
#pragma unroll
  for (int r = 0; r < RANK; ++r)
    for (int off = 16; off; off >>= 1) acc[r] += __shfl_xor(acc[r], off, 32);
  if (lane == 0) {
#pragma unroll
    for (int r = 0; r < RANK; ++r) P[(size_t)t * RANK + r] = acc[r];
  }
}

// Q[t,r] = sum_h H[t,h] * A[r,h]   (H bf16)
__global__ void lora_h_kernel(const unsigned short* __restrict__ Hb,
                              const float* __restrict__ A,
                              float* __restrict__ Q) {
  const int lane = threadIdx.x & 31;
  const int wave = threadIdx.x >> 5;
  const int t = blockIdx.x * 8 + wave;
  float acc[RANK] = {};
  const unsigned short* hr = Hb + (size_t)t * HIDX;
  for (int d = lane; d < HIDX; d += 32) {
    float hv = bf2f(hr[d]);
#pragma unroll
    for (int r = 0; r < RANK; ++r) acc[r] += hv * A[(size_t)r * HIDX + d];
  }
#pragma unroll
  for (int r = 0; r < RANK; ++r)
    for (int off = 16; off; off >>= 1) acc[r] += __shfl_xor(acc[r], off, 32);
  if (lane == 0) {
#pragma unroll
    for (int r = 0; r < RANK; ++r) Q[(size_t)t * RANK + r] = acc[r];
  }
}

// ---------------------------------------------------------------- WMMA loads
__device__ __forceinline__ v16bf load_a16(const unsigned short* __restrict__ base,
                                          int ld, int row, int k, int lane) {
  // 16-bit A layout: lanes 0-15 hold row M=lane, K chunks [0,8)+[16,24);
  // lanes 16-31 hold same rows, K chunks [8,16)+[24,32).
  const int r  = row + (lane & 15);
  const int kk = k + ((lane >> 4) << 3);
  const unsigned short* p = base + (size_t)r * ld + kk;
  v8bf c0 = *(const v8bf*)(p);
  v8bf c1 = *(const v8bf*)(p + 16);
  return __builtin_shufflevector(c0, c1, 0, 1, 2, 3, 4, 5, 6, 7,
                                 8, 9, 10, 11, 12, 13, 14, 15);
}
__device__ __forceinline__ v16bf load_b16(const unsigned short* __restrict__ base,
                                          int ld, int col, int k, int lane) {
  // B (KxN) layout: lane holds column N=lane%16, K chunks [0,8)+[16,24)
  // for lanes 0-15 and [8,16)+[24,32) for lanes 16-31.
  const int c  = col + (lane & 15);
  const int kk = k + ((lane >> 4) << 3);
  const unsigned short* p = base + (size_t)c * ld + kk;
  v8bf c0 = *(const v8bf*)(p);
  v8bf c1 = *(const v8bf*)(p + 16);
  return __builtin_shufflevector(c0, c1, 0, 1, 2, 3, 4, 5, 6, 7,
                                 8, 9, 10, 11, 12, 13, 14, 15);
}

// ---------------------------------------------------------------- GEMM 1
// H = gelu( Xb[N,DIM] * W1b[HID,DIM]^T + P*B1^T + b1 )    (bf16 out)
// 64x64 output tile per wave: 4x4 WMMA accumulators.
__global__ __launch_bounds__(256) void ffn1_kernel(
    const unsigned short* __restrict__ Xb,
    const unsigned short* __restrict__ W1b,
    const float* __restrict__ B1,     // [HID, RANK]
    const float* __restrict__ b1,     // [HID]
    const float* __restrict__ P,      // [N, RANK]
    unsigned short* __restrict__ H) { // [N, HID]
  const int lane = threadIdx.x & 31;
  const int wave = threadIdx.x >> 5;
  const int m0 = blockIdx.x * 64;
  const int n0 = blockIdx.y * 512 + wave * 64;

  v8f acc[4][4] = {};
#pragma unroll 1
  for (int k0 = 0; k0 < DIMX; k0 += 32) {
    v16bf a[4], b[4];
#pragma unroll
    for (int mi = 0; mi < 4; ++mi)
      a[mi] = load_a16(Xb, DIMX, m0 + mi * 16, k0, lane);
#pragma unroll
    for (int ni = 0; ni < 4; ++ni) {
      b[ni] = load_b16(W1b, DIMX, n0 + ni * 16, k0, lane);
      __builtin_prefetch(W1b + (size_t)(n0 + ni * 16 + (lane & 15)) * DIMX +
                             k0 + 32, 0, 1);
    }
#pragma unroll
    for (int mi = 0; mi < 4; ++mi)
#pragma unroll
      for (int ni = 0; ni < 4; ++ni)
        acc[mi][ni] = __builtin_amdgcn_wmma_f32_16x16x32_bf16(
            false, a[mi], false, b[ni], (short)0, acc[mi][ni], false, false);
  }

  // epilogue: C layout — lanes 0-15: vgpr i == row i; lanes 16-31: row 8+i.
  const int nlane = lane & 15;
  const int msel = (lane >> 4) << 3;
#pragma unroll
  for (int mi = 0; mi < 4; ++mi) {
#pragma unroll
    for (int i = 0; i < 8; ++i) {
      const int m = m0 + mi * 16 + msel + i;
      const float* Pm = P + (size_t)m * RANK;
#pragma unroll
      for (int ni = 0; ni < 4; ++ni) {
        const int n = n0 + ni * 16 + nlane;
        float v = acc[mi][ni][i] + b1[n];
        const float* Br = B1 + (size_t)n * RANK;
#pragma unroll
        for (int r = 0; r < RANK; ++r) v += Br[r] * Pm[r];
        v = 0.5f * v * (1.0f + erff(v * 0.70710678118654752f)); // exact gelu
        H[(size_t)m * HIDX + n] = f2bf(v);
      }
    }
  }
}

// ---------------------------------------------------------------- GEMM 2
// y += cw[:,e] * ( Hb[N,HID] * W2b[DIM,HID]^T + Q*B2^T + b2 )
__global__ __launch_bounds__(256) void ffn2_kernel(
    const unsigned short* __restrict__ Hb,
    const unsigned short* __restrict__ W2b,
    const float* __restrict__ B2,     // [DIM, RANK]
    const float* __restrict__ b2,     // [DIM]
    const float* __restrict__ Q,      // [N, RANK]
    const float* __restrict__ cwe,    // cw + e (stride 4)
    float* __restrict__ y) {          // [N, DIM]
  const int lane = threadIdx.x & 31;
  const int wave = threadIdx.x >> 5;
  const int m0 = blockIdx.x * 64;
  const int n0 = blockIdx.y * 512 + wave * 64;

  v8f acc[4][4] = {};
#pragma unroll 1
  for (int k0 = 0; k0 < HIDX; k0 += 32) {
    v16bf a[4], b[4];
#pragma unroll
    for (int mi = 0; mi < 4; ++mi)
      a[mi] = load_a16(Hb, HIDX, m0 + mi * 16, k0, lane);
#pragma unroll
    for (int ni = 0; ni < 4; ++ni) {
      b[ni] = load_b16(W2b, HIDX, n0 + ni * 16, k0, lane);
      __builtin_prefetch(W2b + (size_t)(n0 + ni * 16 + (lane & 15)) * HIDX +
                             k0 + 32, 0, 1);
    }
#pragma unroll
    for (int mi = 0; mi < 4; ++mi)
#pragma unroll
      for (int ni = 0; ni < 4; ++ni)
        acc[mi][ni] = __builtin_amdgcn_wmma_f32_16x16x32_bf16(
            false, a[mi], false, b[ni], (short)0, acc[mi][ni], false, false);
  }

  const int nlane = lane & 15;
  const int msel = (lane >> 4) << 3;
#pragma unroll
  for (int mi = 0; mi < 4; ++mi) {
#pragma unroll
    for (int i = 0; i < 8; ++i) {
      const int m = m0 + mi * 16 + msel + i;
      const float w = cwe[(size_t)m * 4];
      const float* Qm = Q + (size_t)m * RANK;
#pragma unroll
      for (int ni = 0; ni < 4; ++ni) {
        const int n = n0 + ni * 16 + nlane;
        float v = acc[mi][ni][i] + b2[n];
        const float* Br = B2 + (size_t)n * RANK;
#pragma unroll
        for (int r = 0; r < RANK; ++r) v += Br[r] * Qm[r];
        y[(size_t)m * DIMX + n] += w * v;   // experts serialized on stream
      }
    }
  }
}

// ---------------------------------------------------------------- launch
extern "C" void kernel_launch(void* const* d_in, const int* in_sizes, int n_in,
                              void* d_out, int out_size, void* d_ws,
                              size_t ws_size, hipStream_t stream) {
  const float* x  = (const float*)d_in[0];
  const float* Wr = (const float*)d_in[1];
  const float* br = (const float*)d_in[2];
  const float* W1 = (const float*)d_in[3];
  const float* A1 = (const float*)d_in[4];
  const float* B1 = (const float*)d_in[5];
  const float* b1 = (const float*)d_in[6];
  const float* W2 = (const float*)d_in[7];
  const float* A2 = (const float*)d_in[8];
  const float* B2 = (const float*)d_in[9];
  const float* b2 = (const float*)d_in[10];
  float* y = (float*)d_out;

  char* ws = (char*)d_ws;
  size_t off = 0;
  auto alloc = [&](size_t bytes) -> void* {
    void* p = ws + off;
    off = (off + bytes + 255) & ~(size_t)255;
    return p;
  };
  unsigned short* xb  = (unsigned short*)alloc((size_t)NTOK * DIMX * 2);
  unsigned short* w1b = (unsigned short*)alloc((size_t)NEXP * HIDX * DIMX * 2);
  unsigned short* w2b = (unsigned short*)alloc((size_t)NEXP * DIMX * HIDX * 2);
  unsigned short* Hb  = (unsigned short*)alloc((size_t)NTOK * HIDX * 2);
  float* cw = (float*)alloc((size_t)NTOK * NEXP * 4);
  float* P  = (float*)alloc((size_t)NTOK * RANK * 4);
  float* Q  = (float*)alloc((size_t)NTOK * RANK * 4);
  (void)ws_size; (void)in_sizes; (void)n_in; (void)out_size;

  cvt_bf16_kernel<<<4096, 256, 0, stream>>>(x, xb, NTOK * DIMX);
  cvt_bf16_kernel<<<4096, 256, 0, stream>>>(W1, w1b, NEXP * HIDX * DIMX);
  cvt_bf16_kernel<<<4096, 256, 0, stream>>>(W2, w2b, NEXP * DIMX * HIDX);
  router_kernel<<<NTOK / 8, 256, 0, stream>>>(x, Wr, br, cw);
  zero_kernel<<<2048, 256, 0, stream>>>(y, NTOK * DIMX);

  for (int e = 0; e < NEXP; ++e) {
    lora_x_kernel<<<NTOK / 8, 256, 0, stream>>>(
        x, A1 + (size_t)e * RANK * DIMX, P, DIMX);
    ffn1_kernel<<<dim3(NTOK / 64, HIDX / 512), 256, 0, stream>>>(
        xb, w1b + (size_t)e * HIDX * DIMX, B1 + (size_t)e * HIDX * RANK,
        b1 + (size_t)e * HIDX, P, Hb);
    lora_h_kernel<<<NTOK / 8, 256, 0, stream>>>(
        Hb, A2 + (size_t)e * RANK * HIDX, Q);
    ffn2_kernel<<<dim3(NTOK / 64, DIMX / 512), 256, 0, stream>>>(
        Hb, w2b + (size_t)e * DIMX * HIDX, B2 + (size_t)e * DIMX * RANK,
        b2 + (size_t)e * DIMX, Q, cw + e, y);
  }
}
